// SuperGATNetPPI_40656160424576
// MI455X (gfx1250) — matrix operand order repaired and checked by reading, not compiled, not run on hardware
//
#include <hip/hip_runtime.h>
#include <hip/hip_bf16.h>

// ---------------------------------------------------------------------------
// SuperGAT/PPI 3-layer GAT for gfx1250 (MI455X).
// Dense GEMMs via v_wmma_f32_16x16x32_bf16 (wave32, 16x16 tile per wave).
// Edge softmax via order-preserving-uint atomicMax + f32 atomicAdd.
// ---------------------------------------------------------------------------

typedef __attribute__((ext_vector_type(16))) __bf16 v16bf;
typedef __attribute__((ext_vector_type(8)))  __bf16 v8bf;
typedef __attribute__((ext_vector_type(8)))  float  v8f;

#define GAT_NEG_SLOPE 0.2f

static __device__ __forceinline__ __bf16 to_bf16(float f) {
  union { float f; unsigned u; } in; in.f = f;
  unsigned u = in.u;
  unsigned r = u + 0x7fffu + ((u >> 16) & 1u);   // round-to-nearest-even
  union { unsigned short s; __bf16 b; } out;
  out.s = (unsigned short)(r >> 16);
  return out.b;
}

// Order-preserving float <-> uint map (for atomicMax-based segment max).
static __device__ __forceinline__ unsigned fenc(float x) {
  unsigned b = __float_as_uint(x);
  return (b & 0x80000000u) ? ~b : (b | 0x80000000u);
}
static __device__ __forceinline__ float fdec(unsigned k) {
  return (k & 0x80000000u) ? __uint_as_float(k & 0x7fffffffu)
                           : __uint_as_float(~k);
}

static __device__ __forceinline__ float lrelu(float x) {
  return x > 0.f ? x : GAT_NEG_SLOPE * x;
}
static __device__ __forceinline__ float elu(float x) {
  return x > 0.f ? x : expm1f(x);
}

// ---------------------------------------------------------------------------
// generic fill (u32 pattern)
__global__ void fill_u32_kernel(unsigned* p, unsigned v, long n) {
  long i = blockIdx.x * (long)blockDim.x + threadIdx.x;
  if (i < n) p[i] = v;
}

// f32 -> bf16 conversion (activations)
__global__ void f32_to_bf16_kernel(const float* s, __bf16* d, long n) {
  long i = blockIdx.x * (long)blockDim.x + threadIdx.x;
  if (i < n) d[i] = to_bf16(s[i]);
}

// ---------------------------------------------------------------------------
// Pack weight W[K, Ncols] (f32, row-major) into WMMA B-fragment order, bf16,
// zero-padded to Npad = 16*TN columns.
// Read side: lane's v16bf at Bp[((tn*Kblk + kb)*32 + lane)*16], element r*2+j
// holds B[kb*32 + 2r + j + 16*(lane>>4)][tn*16 + (lane&15)].
__global__ void pack_weight_kernel(const float* W, __bf16* Wp,
                                   int K, int Ncols, int Kblk, long total) {
  long id = blockIdx.x * (long)blockDim.x + threadIdx.x;
  if (id >= total) return;
  int j    = (int)(id & 1);
  int r    = (int)((id >> 1) & 7);
  int lane = (int)((id >> 4) & 31);
  long rest = id >> 9;
  int kb = (int)(rest % Kblk);
  int tn = (int)(rest / Kblk);
  int half = lane >> 4;
  int n16  = lane & 15;
  int k    = kb * 32 + 2 * r + j + 16 * half;
  int col  = tn * 16 + n16;
  float v = (k < K && col < Ncols) ? W[(long)k * Ncols + col] : 0.f;
  Wp[id] = to_bf16(v);
}

// ---------------------------------------------------------------------------
// C[M, Npad] (f32) = A[M, K] (bf16 row-major) * Bpacked   — wave per 16x16 tile
__global__ __launch_bounds__(256)
void gemm_wmma_kernel(const __bf16* __restrict__ A,
                      const __bf16* __restrict__ Bp,
                      float* __restrict__ C,
                      int M, int K, int Npad, int Kblk) {
  int wave = (int)(blockIdx.x * 8 + (threadIdx.x >> 5));
  int lane = threadIdx.x & 31;
  int TN = Npad >> 4;
  int Mtiles = (M + 15) >> 4;
  if (wave >= Mtiles * TN) return;           // uniform per wave: EXEC stays full
  int tm = wave / TN;
  int tn = wave % TN;
  int l15  = lane & 15;
  int half = lane >> 4;

  int arowi = tm * 16 + l15;                 // A-fragment row for this lane
  if (arowi >= M) arowi = M - 1;             // clamp (address only, EXEC intact)
  const __bf16* arow  = A + (long)arowi * K;
  const __bf16* bbase = Bp + ((long)tn * Kblk) * 512 + lane * 16;

  v8f acc = {};
  for (int kb = 0; kb < Kblk; ++kb) {
    const __bf16* ap = arow + kb * 32 + 16 * half;
    v8bf alo = *(const v8bf*)(ap);           // K = 16*half .. +8
    v8bf ahi = *(const v8bf*)(ap + 8);       // K = 16*half+8 .. +8
    v16bf a, b;
    b = *(const v16bf*)(bbase + (long)kb * 512);
#pragma unroll
    for (int i = 0; i < 8; ++i) { a[i] = alo[i]; a[i + 8] = ahi[i]; }
    acc = __builtin_amdgcn_wmma_f32_16x16x32_bf16(
        false, a, false, b, (short)0, acc, false, false);
  }

  // C/D layout: VGPR r -> row r + 8*half; lane&15 -> column
  int crow0 = tm * 16 + 8 * half;
  float* cbase = C + (long)crow0 * Npad + tn * 16 + l15;
#pragma unroll
  for (int r = 0; r < 8; ++r) {
    if (crow0 + r < M) cbase[(long)r * Npad] = acc[r];
  }
}

// ---------------------------------------------------------------------------
// alpha_s[n,h] = <xh[n,h,:], a_src[h,:]>,  alpha_d likewise
__global__ void attn_scores_kernel(const float* __restrict__ xh,
                                   const float* __restrict__ asrc,
                                   const float* __restrict__ adst,
                                   float* __restrict__ as, float* __restrict__ ad,
                                   int N, int H, int Cc, int ld) {
  int id = blockIdx.x * blockDim.x + threadIdx.x;
  if (id >= N * H) return;
  int n = id / H, h = id % H;
  const float* xr = xh + (long)n * ld + h * Cc;
  const float* w1 = asrc + h * Cc;
  const float* w2 = adst + h * Cc;
  float s1 = 0.f, s2 = 0.f;
  for (int c = 0; c < Cc; ++c) { float v = xr[c]; s1 += v * w1[c]; s2 += v * w2[c]; }
  as[id] = s1; ad[id] = s2;
}

// segment max over destinations (order-preserving uint atomicMax)
__global__ void edge_max_kernel(const int* __restrict__ src, const int* __restrict__ dst,
                                const float* __restrict__ as, const float* __restrict__ ad,
                                unsigned* __restrict__ mkey, int E, int H) {
  int e = blockIdx.x * blockDim.x + threadIdx.x;
  if (e >= E) return;
  int s = src[e], d = dst[e];
  for (int h = 0; h < H; ++h) {
    float l = lrelu(as[s * H + h] + ad[d * H + h]);
    atomicMax(&mkey[d * H + h], fenc(l));
  }
}

// segment sum of exp(logit - max)
__global__ void edge_denom_kernel(const int* __restrict__ src, const int* __restrict__ dst,
                                  const float* __restrict__ as, const float* __restrict__ ad,
                                  const unsigned* __restrict__ mkey,
                                  float* __restrict__ denom, int E, int H) {
  int e = blockIdx.x * blockDim.x + threadIdx.x;
  if (e >= E) return;
  int s = src[e], d = dst[e];
  for (int h = 0; h < H; ++h) {
    float l = lrelu(as[s * H + h] + ad[d * H + h]);
    atomicAdd(&denom[d * H + h], __expf(l - fdec(mkey[d * H + h])));
  }
}

// out[dst, j] += alpha(e,h) * xh[src, j],  j = h*C + c
__global__ void edge_agg_kernel(const int* __restrict__ src, const int* __restrict__ dst,
                                const float* __restrict__ as, const float* __restrict__ ad,
                                const unsigned* __restrict__ mkey,
                                const float* __restrict__ denom,
                                const float* __restrict__ xh, float* __restrict__ out,
                                int H, int Cc, int ld, long total) {
  long id = blockIdx.x * (long)blockDim.x + threadIdx.x;
  if (id >= total) return;
  int HC = H * Cc;
  int e = (int)(id / HC);
  int j = (int)(id % HC);
  int h = j / Cc;
  int s = src[e], d = dst[e];
  float l  = lrelu(as[s * H + h] + ad[d * H + h]);
  float m  = fdec(mkey[d * H + h]);
  float al = __expf(l - m) / (denom[d * H + h] + 1e-16f);
  atomicAdd(&out[(long)d * ld + j], al * xh[(long)s * ld + j]);
}

// h = elu(gat + bias + lin + linb)      (layer 1)
__global__ void finalize_lin_elu_kernel(const float* __restrict__ g, int ldg,
                                        const float* __restrict__ bias,
                                        const float* __restrict__ lin, int ldl,
                                        const float* __restrict__ linb,
                                        float* __restrict__ out, int N, int F) {
  long id = blockIdx.x * (long)blockDim.x + threadIdx.x;
  if (id >= (long)N * F) return;
  int n = (int)(id / F), j = (int)(id % F);
  float v = g[(long)n * ldg + j] + bias[j] + lin[(long)n * ldl + j] + linb[j];
  out[id] = elu(v);
}

// h = elu(gat + bias + h)               (layer 2, in-place residual)
__global__ void finalize_res_elu_kernel(const float* __restrict__ g, int ldg,
                                        const float* __restrict__ bias,
                                        float* __restrict__ h, int N, int F) {
  long id = blockIdx.x * (long)blockDim.x + threadIdx.x;
  if (id >= (long)N * F) return;
  int n = (int)(id / F), j = (int)(id % F);
  h[id] = elu(g[(long)n * ldg + j] + bias[j] + h[id]);
}

// out = mean_h(gat) + b3 + lin3 + lin3b (layer 3)
__global__ void finalize3_kernel(const float* __restrict__ g, int ldg,
                                 const float* __restrict__ b3,
                                 const float* __restrict__ lin, int ldl,
                                 const float* __restrict__ lin3b,
                                 float* __restrict__ out, int N, int Cc, int H) {
  long id = blockIdx.x * (long)blockDim.x + threadIdx.x;
  if (id >= (long)N * Cc) return;
  int n = (int)(id / Cc), c = (int)(id % Cc);
  float s = 0.f;
  for (int h = 0; h < H; ++h) s += g[(long)n * ldg + h * Cc + c];
  out[id] = s * (1.f / H) + b3[c] + lin[(long)n * ldl + c] + lin3b[c];
}

// ---------------------------------------------------------------------------
extern "C" void kernel_launch(void* const* d_in, const int* in_sizes, int n_in,
                              void* d_out, int out_size, void* d_ws, size_t ws_size,
                              hipStream_t stream) {
  const int FIN = 128, H = 4, HID = 64, NCLS = 121;
  const int NP12 = H * HID;        // 256
  const int NP3  = 496;            // 484 padded to 16
  const int NPL3 = 128;            // 121 padded to 16
  const int N = in_sizes[0] / FIN;
  const int E = in_sizes[1] / 2;

  const float* x     = (const float*)d_in[0];
  const int*   ei    = (const int*)d_in[1];
  const int*   esrc  = ei;
  const int*   edst  = ei + E;
  const float* W1    = (const float*)d_in[2];
  const float* a1s   = (const float*)d_in[3];
  const float* a1d   = (const float*)d_in[4];
  const float* b1    = (const float*)d_in[5];
  const float* l1W   = (const float*)d_in[6];
  const float* l1b   = (const float*)d_in[7];
  const float* W2    = (const float*)d_in[8];
  const float* a2s   = (const float*)d_in[9];
  const float* a2d   = (const float*)d_in[10];
  const float* b2    = (const float*)d_in[11];
  const float* W3    = (const float*)d_in[12];
  const float* a3s   = (const float*)d_in[13];
  const float* a3d   = (const float*)d_in[14];
  const float* b3    = (const float*)d_in[15];
  const float* l3W   = (const float*)d_in[16];
  const float* l3b   = (const float*)d_in[17];
  float* out = (float*)d_out;
  (void)n_in; (void)out_size; (void)ws_size;

  // ---- workspace carve (256B aligned) ----
  char* wsb = (char*)d_ws; size_t off = 0;
  auto carve = [&](size_t bytes) -> void* {
    void* p = wsb + off; off += (bytes + 255) & ~(size_t)255; return p;
  };
  __bf16* XB   = (__bf16*)carve((size_t)N * 256 * 2);        // bf16 activations
  __bf16* WP1  = (__bf16*)carve((size_t)128 * NP12 * 2);
  __bf16* WPL1 = (__bf16*)carve((size_t)128 * NP12 * 2);
  __bf16* WP2  = (__bf16*)carve((size_t)256 * NP12 * 2);
  __bf16* WP3  = (__bf16*)carve((size_t)256 * NP3 * 2);
  __bf16* WPL3 = (__bf16*)carve((size_t)256 * NPL3 * 2);
  float*  XH   = (float*)carve((size_t)N * NP3 * 4);         // xh (max width)
  float*  GOUT = (float*)carve((size_t)N * NP3 * 4);         // gat accumulation
  float*  LIN  = (float*)carve((size_t)N * 256 * 4);         // linear skip
  float*  HB   = (float*)carve((size_t)N * 256 * 4);         // activations f32
  float*     AS  = (float*)carve((size_t)N * H * 4);
  float*     AD  = (float*)carve((size_t)N * H * 4);
  unsigned*  MK  = (unsigned*)carve((size_t)N * H * 4);
  float*     DEN = (float*)carve((size_t)N * H * 4);

  auto gblk = [](long n, int b) { return (unsigned)((n + b - 1) / b); };

  // ---- pack all weights (f32 -> bf16 WMMA fragment order, zero-padded) ----
  {
    long t1 = (long)128 * NP12;   // == TN*Kblk*512
    pack_weight_kernel<<<gblk(t1, 256), 256, 0, stream>>>(W1, WP1, 128, NP12, 4, t1);
    pack_weight_kernel<<<gblk(t1, 256), 256, 0, stream>>>(l1W, WPL1, 128, NP12, 4, t1);
    long t2 = (long)256 * NP12;
    pack_weight_kernel<<<gblk(t2, 256), 256, 0, stream>>>(W2, WP2, 256, NP12, 8, t2);
    long t3 = (long)256 * NP3;
    pack_weight_kernel<<<gblk(t3, 256), 256, 0, stream>>>(W3, WP3, 256, 484, 8, t3);
    long t4 = (long)256 * NPL3;
    pack_weight_kernel<<<gblk(t4, 256), 256, 0, stream>>>(l3W, WPL3, 256, 121, 8, t4);
  }

  int Mtiles = (N + 15) / 16;
  auto gemm = [&](const __bf16* A, const __bf16* Bp, float* C,
                  int K, int Npad, int Kblk) {
    long tiles = (long)Mtiles * (Npad >> 4);
    gemm_wmma_kernel<<<gblk(tiles, 8), 256, 0, stream>>>(A, Bp, C, N, K, Npad, Kblk);
  };
  auto edge_phase = [&](int Cc, int ld) {
    fill_u32_kernel<<<gblk((long)N * H, 256), 256, 0, stream>>>(MK, 0u, (long)N * H);
    fill_u32_kernel<<<gblk((long)N * H, 256), 256, 0, stream>>>((unsigned*)DEN, 0u, (long)N * H);
    fill_u32_kernel<<<gblk((long)N * ld, 256), 256, 0, stream>>>((unsigned*)GOUT, 0u, (long)N * ld);
    edge_max_kernel<<<gblk(E, 256), 256, 0, stream>>>(esrc, edst, AS, AD, MK, E, H);
    edge_denom_kernel<<<gblk(E, 256), 256, 0, stream>>>(esrc, edst, AS, AD, MK, DEN, E, H);
    long tot = (long)E * H * Cc;
    edge_agg_kernel<<<gblk(tot, 256), 256, 0, stream>>>(esrc, edst, AS, AD, MK, DEN,
                                                        XH, GOUT, H, Cc, ld, tot);
  };

  // ================= layer 1: F_IN -> H*HID (concat) + lin skip =============
  f32_to_bf16_kernel<<<gblk((long)N * FIN, 256), 256, 0, stream>>>(x, XB, (long)N * FIN);
  gemm(XB, WP1, XH, 128, NP12, 4);
  gemm(XB, WPL1, LIN, 128, NP12, 4);
  attn_scores_kernel<<<gblk((long)N * H, 256), 256, 0, stream>>>(XH, a1s, a1d, AS, AD, N, H, HID, NP12);
  edge_phase(HID, NP12);
  finalize_lin_elu_kernel<<<gblk((long)N * NP12, 256), 256, 0, stream>>>(
      GOUT, NP12, b1, LIN, NP12, l1b, HB, N, NP12);

  // ================= layer 2: H*HID -> H*HID (concat) + residual ============
  f32_to_bf16_kernel<<<gblk((long)N * NP12, 256), 256, 0, stream>>>(HB, XB, (long)N * NP12);
  gemm(XB, WP2, XH, 256, NP12, 8);
  attn_scores_kernel<<<gblk((long)N * H, 256), 256, 0, stream>>>(XH, a2s, a2d, AS, AD, N, H, HID, NP12);
  edge_phase(HID, NP12);
  finalize_res_elu_kernel<<<gblk((long)N * NP12, 256), 256, 0, stream>>>(
      GOUT, NP12, b2, HB, N, NP12);

  // ================= layer 3: H*HID -> N_CLS (mean heads) + lin skip ========
  f32_to_bf16_kernel<<<gblk((long)N * NP12, 256), 256, 0, stream>>>(HB, XB, (long)N * NP12);
  gemm(XB, WP3, XH, 256, NP3, 8);
  gemm(XB, WPL3, LIN, 256, NPL3, 8);
  attn_scores_kernel<<<gblk((long)N * H, 256), 256, 0, stream>>>(XH, a3s, a3d, AS, AD, N, H, NCLS, NP3);
  edge_phase(NCLS, NP3);
  finalize3_kernel<<<gblk((long)N * NCLS, 256), 256, 0, stream>>>(
      GOUT, NP3, b3, LIN, NPL3, l3b, out, N, NCLS, H);
}
